// MultiHeadSelfAttention_84189948936447
// MI455X (gfx1250) — matrix operand order repaired and checked by reading, not compiled, not run on hardware
//
#include <hip/hip_runtime.h>
#include <hip/hip_bf16.h>

// Multi-head self-attention, B=2 S=2048 D=1024 H=16 Hd=64, fp32 in/out.
// All matmuls on v_wmma_f32_16x16x32_bf16 (fp32 accum). Flash attention
// stages K/V tiles in LDS via global_load_async_to_lds_b128 (ASYNCcnt).

typedef __bf16 bf16;
typedef __attribute__((ext_vector_type(16))) __bf16 bf16x16;
typedef __attribute__((ext_vector_type(8)))  __bf16 bf16x8;
typedef __attribute__((ext_vector_type(8)))  float  f32x8;

union FragAB { bf16x16 v; bf16x8 h[2]; };

// ---- WMMA fragment loaders (wave32 layouts per CDNA5 ISA 7.12.2) ----------
// A (16x32, 16-bit): lane L -> row m=L&15, elems e0..7 = K(kb..kb+7),
// e8..15 = K(kb+16..kb+23), kb = (L>=16)*8. Two contiguous 16B loads.
__device__ __forceinline__ bf16x16 load_a_frag(const bf16* base, int ld,
                                               int m0, int k0, int lane) {
  int m  = m0 + (lane & 15);
  int kb = k0 + ((lane >> 4) << 3);
  const bf16x8* p = (const bf16x8*)(base + (size_t)m * ld + kb);
  FragAB f; f.h[0] = p[0]; f.h[1] = p[2];   // +0 elems, +16 elems
  return f.v;
}
// B (32x16, 16-bit): element (k,n) at base[n*ld + k]. lane L -> col n=L&15,
// e0..7 = K(kb..kb+7), e8..15 = K(kb+16..kb+23), kb = (L>=16)*8. Two 16B loads.
__device__ __forceinline__ bf16x16 load_b_frag(const bf16* base, int ld,
                                               int n0, int k0, int lane) {
  int n  = n0 + (lane & 15);
  int kb = k0 + ((lane >> 4) << 3);
  const bf16x8* p = (const bf16x8*)(base + (size_t)n * ld + kb);
  FragAB f; f.h[0] = p[0]; f.h[1] = p[2];
  return f.v;
}

__device__ __forceinline__ f32x8 wmma_bf16(bf16x16 a, bf16x16 b, f32x8 c) {
  return __builtin_amdgcn_wmma_f32_16x16x32_bf16(false, a, false, b,
                                                 (short)0, c, false, false);
}

// ---- CDNA5 async memory->LDS (tracked by ASYNCcnt, CDNA5 ISA) ----------
__device__ __forceinline__ void async_b128(const bf16* g, const bf16* l) {
  unsigned lofs = (unsigned)(uintptr_t)l;   // flat LDS addr low 32 bits = offset
  asm volatile("global_load_async_to_lds_b128 %0, %1, off"
               :: "v"(lofs), "v"(g) : "memory");
}
__device__ __forceinline__ void wait_async_le(int keep4) {
  if (keep4) asm volatile("s_wait_asynccnt 0x4" ::: "memory");
  else       asm volatile("s_wait_asynccnt 0x0" ::: "memory");
}

// ---- fp32 -> bf16 convert --------------------------------------------------
__global__ void cvt_f32_bf16(const float* __restrict__ in,
                             bf16* __restrict__ out, int n) {
  int i = blockIdx.x * blockDim.x + threadIdx.x;
  int stride = gridDim.x * blockDim.x;
  for (; i < n; i += stride) out[i] = (bf16)in[i];
}

// ---- QKV projection: qkv = x @ w_qkv^T + b_qkv ----------------------------
// Wave tile 32x64 (8 WMMA / 12 b128 loads per K=32 step); block 128x128.
// Epilogue scatters into WMMA-native layouts: Q [B,H,S,64]*(1/8),
// K [B,H,S,64], V^T [B,H,64,S].
__global__ __launch_bounds__(256) void qkv_gemm(
    const bf16* __restrict__ x, const bf16* __restrict__ w,
    const float* __restrict__ bias,
    bf16* __restrict__ qbuf, bf16* __restrict__ kbuf, bf16* __restrict__ vtbuf) {
  const int lane = threadIdx.x & 31;
  const int wave = threadIdx.x >> 5;                     // 8 waves: 4 rows x 2 cols
  const int m0 = blockIdx.x * 128 + (wave >> 1) * 32;
  const int n0 = blockIdx.y * 128 + (wave & 1) * 64;

  f32x8 acc[2][4] = {};
  for (int k0 = 0; k0 < 1024; k0 += 32) {
    bf16x16 a0 = load_a_frag(x, 1024, m0,      k0, lane);
    bf16x16 a1 = load_a_frag(x, 1024, m0 + 16, k0, lane);
    bf16x16 bj[4];
#pragma unroll
    for (int j = 0; j < 4; ++j) bj[j] = load_b_frag(w, 1024, n0 + j * 16, k0, lane);
#pragma unroll
    for (int j = 0; j < 4; ++j) {
      acc[0][j] = wmma_bf16(a0, bj[j], acc[0][j]);
      acc[1][j] = wmma_bf16(a1, bj[j], acc[1][j]);
    }
  }

#pragma unroll
  for (int j = 0; j < 4; ++j) {
    const int n  = n0 + j * 16 + (lane & 15);
    const float bn = bias[n];
    const int which = n >> 10;        // 0=q 1=k 2=v (uniform per 16-col subtile)
    const int d  = n & 1023;
    const int h  = d >> 6;            // uniform per subtile
    const int hd = d & 63;
#pragma unroll
    for (int i = 0; i < 2; ++i) {
      const int mb = m0 + i * 16 + ((lane >> 4) << 3);
#pragma unroll
      for (int r = 0; r < 8; ++r) {
        const int m = mb + r;
        const int bb = m >> 11, s = m & 2047;
        const float v = acc[i][j][r] + bn;
        const size_t hslot = (size_t)(bb * 16 + h);
        if (which == 0)      qbuf [(hslot * 2048 + s) * 64 + hd] = (bf16)(v * 0.125f);
        else if (which == 1) kbuf [(hslot * 2048 + s) * 64 + hd] = (bf16)v;
        else                 vtbuf[(hslot * 64 + hd) * 2048 + s] = (bf16)v;
      }
    }
  }
}

// ---- Flash attention -------------------------------------------------------
// Block = 64 query rows (4 waves x 16), one (b,h). K/V tiles (32 keys) staged
// in LDS once per block via async-to-LDS, triple-buffered (single barrier per
// tile is race-free: writes land 2 tiles ahead of any reader).
#define KSTRIDE 72   // 32x64 K tile rows padded: 144B = 36 banks, conflict-free
#define VSTRIDE 40   // 64x32 V^T tile rows padded: 80B = 20 banks, conflict-free
__global__ __launch_bounds__(128) void flash_attn(
    const bf16* __restrict__ qbuf, const bf16* __restrict__ kbuf,
    const bf16* __restrict__ vtbuf, bf16* __restrict__ attn /*[2,2048,1024]*/) {
  __shared__ bf16 sK[3][32 * KSTRIDE];
  __shared__ bf16 sV[3][64 * VSTRIDE];
  __shared__ bf16 sP[4][16 * 40];
  const int tid  = threadIdx.x;
  const int lane = tid & 31;
  const int wave = tid >> 5;
  const int h = blockIdx.y, b = blockIdx.z;
  const int q0 = blockIdx.x * 64 + wave * 16;

  const bf16* qh = qbuf  + (size_t)(b * 16 + h) * 2048 * 64;
  const bf16* kh = kbuf  + (size_t)(b * 16 + h) * 2048 * 64;
  const bf16* vh = vtbuf + (size_t)(b * 16 + h) * 64 * 2048;

  // Q fragments live in registers all kernel (pre-scaled by 1/sqrt(Hd))
  const bf16x16 qa0 = load_a_frag(qh, 64, q0, 0,  lane);
  const bf16x16 qa1 = load_a_frag(qh, 64, q0, 32, lane);

  f32x8 o[4] = {{}, {}, {}, {}};
  float rmax[8], rsum[8];
#pragma unroll
  for (int r = 0; r < 8; ++r) { rmax[r] = -3.0e38f; rsum[r] = 0.0f; }
  bf16* P = sP[wave];

  // cooperative async stage of one 32-key K/V tile (4 async ops per thread)
  auto stage = [&](int kt, int buf) {
#pragma unroll
    for (int c = tid; c < 256; c += 128) {        // K: 32 keys x 8 chunks
      int key = c >> 3, cc = c & 7;
      async_b128(kh + (size_t)(kt + key) * 64 + cc * 8,
                 &sK[buf][key * KSTRIDE + cc * 8]);
    }
#pragma unroll
    for (int c = tid; c < 256; c += 128) {        // V^T: 64 hd rows x 4 chunks
      int hd = c >> 2, cc = c & 3;
      async_b128(vh + (size_t)hd * 2048 + kt + cc * 8,
                 &sV[buf][hd * VSTRIDE + cc * 8]);
    }
  };

  stage(0, 0);
  for (int it = 0; it < 64; ++it) {
    const int prefetch = (it + 1 < 64);
    if (prefetch) stage((it + 1) * 32, (it + 1) % 3);
    wait_async_le(prefetch);          // current tile's 4 ops retired
    __syncthreads();
    const bf16* Kb = sK[it % 3];
    const bf16* Vb = sV[it % 3];

    // scores: two 16-key sub-tiles, Hd=64 -> two K=32 WMMA steps each
    f32x8 s0 = {}, s1 = {};
    s0 = wmma_bf16(qa0, load_b_frag(Kb, KSTRIDE, 0,  0,  lane), s0);
    s0 = wmma_bf16(qa1, load_b_frag(Kb, KSTRIDE, 0,  32, lane), s0);
    s1 = wmma_bf16(qa0, load_b_frag(Kb, KSTRIDE, 16, 0,  lane), s1);
    s1 = wmma_bf16(qa1, load_b_frag(Kb, KSTRIDE, 16, 32, lane), s1);

    // online softmax (C-fragment row r lives across one 16-lane half)
    float c[8];
#pragma unroll
    for (int r = 0; r < 8; ++r) {
      float t = fmaxf(s0[r], s1[r]);
      t = fmaxf(t, __shfl_xor(t, 1));
      t = fmaxf(t, __shfl_xor(t, 2));
      t = fmaxf(t, __shfl_xor(t, 4));
      t = fmaxf(t, __shfl_xor(t, 8));
      const float nm = fmaxf(rmax[r], t);
      c[r] = __expf(rmax[r] - nm);
      rmax[r] = nm;
      const float p0 = __expf(s0[r] - nm);
      const float p1 = __expf(s1[r] - nm);
      s0[r] = p0; s1[r] = p1;
      float ps = p0 + p1;
      ps += __shfl_xor(ps, 1);
      ps += __shfl_xor(ps, 2);
      ps += __shfl_xor(ps, 4);
      ps += __shfl_xor(ps, 8);
      rsum[r] = rsum[r] * c[r] + ps;
    }

    // C-layout -> A-layout transpose of P via wave-private LDS (DS in-order)
    {
      const int col = lane & 15;
      const int rb  = (lane >> 4) << 3;
#pragma unroll
      for (int r = 0; r < 8; ++r) {
        bf16* row = P + (rb + r) * 40;
        row[col]      = (bf16)s0[r];
        row[col + 16] = (bf16)s1[r];
      }
    }
    bf16x16 pa;
    {
      const int m  = lane & 15;
      const int kb = (lane >> 4) << 3;
      FragAB f;
      f.h[0] = *(const bf16x8*)(P + m * 40 + kb);
      f.h[1] = *(const bf16x8*)(P + m * 40 + kb + 16);
      pa = f.v;
    }

    // rescale running output then P @ V
#pragma unroll
    for (int t = 0; t < 4; ++t) {
      f32x8 ot = o[t];
#pragma unroll
      for (int r = 0; r < 8; ++r) ot[r] *= c[r];
      o[t] = wmma_bf16(pa, load_b_frag(Vb, VSTRIDE, t * 16, 0, lane), ot);
    }
    __syncthreads();   // (cheap insurance: readers done before far-future reuse)
  }

  // normalize + store attn [B,S,D] bf16
  {
    const int n  = lane & 15;
    const int mb = q0 + ((lane >> 4) << 3);
#pragma unroll
    for (int r = 0; r < 8; ++r) {
      const float inv = 1.0f / rsum[r];
      bf16* dst = attn + ((size_t)b * 2048 + (mb + r)) * 1024 + h * 64 + n;
#pragma unroll
      for (int t = 0; t < 4; ++t) dst[t * 16] = (bf16)(o[t][r] * inv);
    }
  }
}

// ---- Output projection: out = attn @ w_out^T + b_out (fp32 out) ------------
__global__ __launch_bounds__(256) void out_gemm(
    const bf16* __restrict__ attn, const bf16* __restrict__ w,
    const float* __restrict__ bias, float* __restrict__ out) {
  const int lane = threadIdx.x & 31;
  const int wave = threadIdx.x >> 5;
  const int m0 = blockIdx.x * 128 + (wave >> 1) * 32;
  const int n0 = blockIdx.y * 128 + (wave & 1) * 64;

  f32x8 acc[2][4] = {};
  for (int k0 = 0; k0 < 1024; k0 += 32) {
    bf16x16 a0 = load_a_frag(attn, 1024, m0,      k0, lane);
    bf16x16 a1 = load_a_frag(attn, 1024, m0 + 16, k0, lane);
    bf16x16 bj[4];
#pragma unroll
    for (int j = 0; j < 4; ++j) bj[j] = load_b_frag(w, 1024, n0 + j * 16, k0, lane);
#pragma unroll
    for (int j = 0; j < 4; ++j) {
      acc[0][j] = wmma_bf16(a0, bj[j], acc[0][j]);
      acc[1][j] = wmma_bf16(a1, bj[j], acc[1][j]);
    }
  }
#pragma unroll
  for (int j = 0; j < 4; ++j) {
    const int n = n0 + j * 16 + (lane & 15);
    const float bn = bias[n];
#pragma unroll
    for (int i = 0; i < 2; ++i) {
      const int mb = m0 + i * 16 + ((lane >> 4) << 3);
#pragma unroll
      for (int r = 0; r < 8; ++r)
        out[(size_t)(mb + r) * 1024 + n] = acc[i][j][r] + bn;
    }
  }
}

// ---- launch ----------------------------------------------------------------
extern "C" void kernel_launch(void* const* d_in, const int* in_sizes, int n_in,
                              void* d_out, int out_size, void* d_ws, size_t ws_size,
                              hipStream_t stream) {
  const float* x     = (const float*)d_in[0];   // [2,2048,1024]
  const float* w_qkv = (const float*)d_in[1];   // [3072,1024]
  const float* b_qkv = (const float*)d_in[2];   // [3072]
  const float* w_out = (const float*)d_in[3];   // [1024,1024]
  const float* b_out = (const float*)d_in[4];   // [1024]
  float* out = (float*)d_out;                   // [2,2048,1024]

  // Workspace layout (needs 48 MB)
  char* p = (char*)d_ws;
  bf16* x_bf    = (bf16*)p; p += (size_t)4096 * 1024 * 2;
  bf16* wqkv_bf = (bf16*)p; p += (size_t)3072 * 1024 * 2;
  bf16* wout_bf = (bf16*)p; p += (size_t)1024 * 1024 * 2;
  bf16* qbuf    = (bf16*)p; p += (size_t)2 * 16 * 2048 * 64 * 2;
  bf16* kbuf    = (bf16*)p; p += (size_t)2 * 16 * 2048 * 64 * 2;
  bf16* vtbuf   = (bf16*)p; p += (size_t)2 * 16 * 2048 * 64 * 2;
  bf16* attn    = (bf16*)p;

  cvt_f32_bf16<<<2048, 256, 0, stream>>>(x,     x_bf,    4096 * 1024);
  cvt_f32_bf16<<<2048, 256, 0, stream>>>(w_qkv, wqkv_bf, 3072 * 1024);
  cvt_f32_bf16<<<1024, 256, 0, stream>>>(w_out, wout_bf, 1024 * 1024);

  qkv_gemm<<<dim3(32, 24), 256, 0, stream>>>(x_bf, wqkv_bf, b_qkv,
                                             qbuf, kbuf, vtbuf);
  flash_attn<<<dim3(32, 16, 2), 128, 0, stream>>>(qbuf, kbuf, vtbuf, attn);
  out_gemm<<<dim3(32, 8), 256, 0, stream>>>(attn, wout_bf, b_out, out);
}
